// TokenStage_79998060855893
// MI455X (gfx1250) — compile-verified, matrix operand order checked by reading, not run on hardware
//
#include <hip/hip_runtime.h>
#include <cstdint>
#include <cstddef>

// ---------------- types ----------------
typedef __attribute__((ext_vector_type(16))) __bf16 v16bf;
typedef __attribute__((ext_vector_type(8)))  float  v8f;
typedef __attribute__((ext_vector_type(4)))  int    v4i_;
typedef __attribute__((address_space(1))) v4i_* gv4i_p;   // global int4*
typedef __attribute__((address_space(3))) v4i_* lv4i_p;   // LDS int4*

#define IMGSZ 256
#define GRD   64
#define NTOK  4096
#define BATCH 2
#define CH    192
#define ROWS  (BATCH*NTOK)   /* 8192 */
#define DI_   384
#define DSTATE 16
#define DTRK  12
#define NHEADS 6
#define HDIM  32
#define WSZ   8
#define MLPH_ 768

// gfx1250 async global->LDS path (guarded; falls back to sync LDS copy)
#if defined(__has_builtin)
#  if __has_builtin(__builtin_amdgcn_global_load_async_to_lds_b128) && \
      __has_builtin(__builtin_amdgcn_s_wait_asynccnt)
#    define HAVE_ASYNC_LDS 1
#  endif
#endif
#ifndef HAVE_ASYNC_LDS
#  define HAVE_ASYNC_LDS 0
#endif

__device__ __forceinline__ float geluf(float x){ return 0.5f*x*(1.0f+erff(x*0.70710678118f)); }
__device__ __forceinline__ float softplusf(float x){ return (x>20.f)?x:log1pf(expf(x)); }
__device__ __forceinline__ float siluf(float x){ return x/(1.0f+expf(-x)); }

// =====================================================================
// WMMA GEMM:  D[M,N] = A[M,K](bf16) * W[N,K](bf16)^T + bias
// Block = 128 threads = 4 waves sharing one A row-strip (16 rows).
// A K-slab (16x32 bf16) is staged into LDS (double buffered) by wave 0
// using global_load_async_to_lds_b128 (ASYNCcnt) when available.
// Each wave owns one 16x16 output tile; W fragments are direct b128 loads.
// =====================================================================
__global__ void gemm_bf16_wmma_kernel(const __bf16* __restrict__ A,
                                      const __bf16* __restrict__ W,
                                      const float*  __restrict__ bias,
                                      float* __restrict__ D,
                                      int M, int N, int K, int act, int nStrips)
{
  __shared__ __bf16 sA[2*16*32];              // double-buffered A slab
  int tilesN = N >> 4;
  int strip = blockIdx.x % nStrips;
  int tm    = blockIdx.x / nStrips;
  int tid = (int)threadIdx.x;
  int w = tid >> 5, lane = tid & 31, half = lane >> 4, r = lane & 15;

  int tn = strip*4 + w;
  bool active = (tn < tilesN);
  if (!active) tn = tilesN - 1;               // clamp: stay converged, skip store

  const __bf16* wrow = W + (size_t)(tn*16 + r) * K + half*8;

  auto stage = [&](int buf, int k0) {         // wave-0 lanes only
#pragma unroll
    for (int T = lane; T < 64; T += 32) {
      int row = T >> 2, seg = T & 3;          // 16 rows x 4 x 8bf16 segments
      const __bf16* g = A + (size_t)(tm*16 + row)*K + k0 + seg*8;
      __bf16* l = sA + buf*512 + row*32 + seg*8;
#if HAVE_ASYNC_LDS
      __builtin_amdgcn_global_load_async_to_lds_b128((gv4i_p)g, (lv4i_p)l, 0, 0);
#else
      *(uint4*)l = *(const uint4*)g;
#endif
    }
  };

  if (w == 0) stage(0, 0);

  v8f acc = {};
  int buf = 0;
  for (int k0 = 0; k0 < K; k0 += 32, buf ^= 1) {
#if HAVE_ASYNC_LDS
    if (w == 0) __builtin_amdgcn_s_wait_asynccnt(0);
#endif
    __syncthreads();
    if (w == 0 && k0 + 32 < K) stage(buf ^ 1, k0 + 32);
    if (k0 + 32 < K) __builtin_prefetch((const void*)(wrow + k0 + 32), 0, 1);

    union { v16bf v; uint4 u[2]; } fa, fb;
    const __bf16* ar = sA + buf*512 + r*32 + half*8;
    fa.u[0] = *(const uint4*)(ar);
    fa.u[1] = *(const uint4*)(ar + 16);
    fb.u[0] = *(const uint4*)(wrow + k0);
    fb.u[1] = *(const uint4*)(wrow + k0 + 16);
    acc = __builtin_amdgcn_wmma_f32_16x16x32_bf16(false, fa.v, false, fb.v,
                                                  (short)0, acc, false, false);
  }

  if (active) {
    int col = tn*16 + r;
    float bv = bias ? bias[col] : 0.f;
#pragma unroll
    for (int i = 0; i < 8; i++) {
      int row = tm*16 + i + half*8;
      float v = acc[i] + bv;
      if (act == 1) v = geluf(v);
      else if (act == 2) v = softplusf(v);
      D[(size_t)row * N + col] = v;
    }
  }
}

// =====================================================================
// 5x5 gaussian depthwise blur, low = blur * w, high = x - low
// =====================================================================
__global__ void blur_split_kernel(const float* __restrict__ x,
                                  const float* __restrict__ blur_w,
                                  float* __restrict__ low, float* __restrict__ high)
{
  int id = blockIdx.x * blockDim.x + threadIdx.x;
  if (id >= BATCH*3*IMGSZ*IMGSZ) return;
  int j = id & 255, i = (id >> 8) & 255, bc = id >> 16;
  const float* img = x + (size_t)bc * IMGSZ * IMGSZ;
  const float inv2s2 = 1.0f / (2.0f * (5.0f/6.0f) * (5.0f/6.0f));
  float acc = 0.f, wsum = 0.f;
#pragma unroll
  for (int dy = -2; dy <= 2; dy++)
#pragma unroll
    for (int dx = -2; dx <= 2; dx++) {
      float wv = expf(-(float)(dy*dy + dx*dx) * inv2s2);
      wsum += wv;
      int yy = i + dy, xx = j + dx;
      float pv = (yy >= 0 && yy < IMGSZ && xx >= 0 && xx < IMGSZ) ? img[yy*IMGSZ + xx] : 0.f;
      acc += wv * pv;
    }
  float lo = acc / wsum * blur_w[0];
  low[id] = lo;
  high[id] = x[id] - lo;
}

// im2col for patch embed: A[8192][64] (K=48 valid, padded to 64)
__global__ void im2col_embed_kernel(const float* __restrict__ img, __bf16* __restrict__ A)
{
  int id = blockIdx.x * blockDim.x + threadIdx.x;
  if (id >= ROWS*64) return;
  int k = id & 63, m = id >> 6;
  if (k >= 48) { A[id] = (__bf16)0.f; return; }
  int c = k >> 4, py = (k >> 2) & 3, px = k & 3;
  int b = m >> 12, pos = m & 4095, gy = pos >> 6, gx = pos & 63;
  A[id] = (__bf16)img[(((size_t)b*3 + c)*IMGSZ + gy*4 + py)*IMGSZ + gx*4 + px];
}

// generic f32 -> bf16 with 2D padding
__global__ void convert_pad_kernel(const float* __restrict__ src, int src_ld, int Nsrc, int Ksrc,
                                   __bf16* __restrict__ dst, int Ndst, int Kdst)
{
  int id = blockIdx.x * blockDim.x + threadIdx.x;
  if (id >= Ndst * Kdst) return;
  int k = id % Kdst, n = id / Kdst;
  float v = (n < Nsrc && k < Ksrc) ? src[(size_t)n * src_ld + k] : 0.f;
  dst[id] = (__bf16)v;
}

// LayerNorm: one wave per row
__global__ void ln_rows_kernel(const float* __restrict__ x, const float* __restrict__ g,
                               const float* __restrict__ b, float* __restrict__ y,
                               int rows, int cols)
{
  int wid = (int)((blockIdx.x * blockDim.x + threadIdx.x) >> 5);
  int lane = threadIdx.x & 31;
  if (wid >= rows) return;
  const float* xr = x + (size_t)wid * cols;
  float s = 0.f, s2 = 0.f;
  for (int c = lane; c < cols; c += 32) { float v = xr[c]; s += v; s2 += v*v; }
#pragma unroll
  for (int m = 16; m; m >>= 1) { s += __shfl_xor(s, m); s2 += __shfl_xor(s2, m); }
  float mean = s / cols;
  float var = s2 / cols - mean*mean;
  float inv = rsqrtf(var + 1e-5f);
  float* yr = y + (size_t)wid * cols;
  for (int c = lane; c < cols; c += 32) yr[c] = (xr[c]-mean)*inv*g[c] + b[c];
}

// causal depthwise conv1d (DCONV=4) + silu
__global__ void conv1d_silu_kernel(const float* __restrict__ xz, const float* __restrict__ cw,
                                   const float* __restrict__ cb, float* __restrict__ xc)
{
  int id = blockIdx.x * blockDim.x + threadIdx.x;
  if (id >= ROWS*DI_) return;
  int d = id % DI_;
  int m = id / DI_;
  int t = m & 4095, b = m >> 12;
  float acc = cb[d];
#pragma unroll
  for (int j = 0; j < 4; j++) {
    int tt = t - 3 + j;
    if (tt >= 0) acc += cw[d*4 + j] * xz[((size_t)(b*NTOK + tt))*768 + d];
  }
  xc[id] = siluf(acc);
}

// selective-scan: 16 lanes per (b,d), shuffle-reduce the 16 states each step
__global__ void mamba_scan_kernel(const float* __restrict__ delta, const float* __restrict__ dbl,
                                  const float* __restrict__ xc, const float* __restrict__ A_log,
                                  float* __restrict__ ysc)
{
  int gid = blockIdx.x * blockDim.x + threadIdx.x;
  int grp = gid >> 4;
  int s = gid & 15;
  if (grp >= BATCH*DI_) return;
  int b = grp / DI_, d = grp % DI_;
  float a = -expf(A_log[d*DSTATE + s]);
  float h = 0.f;
  size_t base = (size_t)b * NTOK;
  for (int t = 0; t < NTOK; t++) {
    size_t row = base + t;
    float dv = delta[row*DI_ + d];
    float xv = xc[row*DI_ + d];
    float Bv = dbl[row*48 + 12 + s];
    float Cv = dbl[row*48 + 28 + s];
    h = expf(dv * a) * h + dv * Bv * xv;
    float c = h * Cv;
    c += __shfl_xor(c, 1); c += __shfl_xor(c, 2);
    c += __shfl_xor(c, 4); c += __shfl_xor(c, 8);
    if (s == 0) ysc[row*DI_ + d] = c;
  }
}

// y = (ysc + xc*D) * silu(z) -> bf16
__global__ void mamba_gate_kernel(const float* __restrict__ ysc, const float* __restrict__ xc,
                                  const float* __restrict__ xz, const float* __restrict__ Dp,
                                  __bf16* __restrict__ Abf)
{
  int id = blockIdx.x * blockDim.x + threadIdx.x;
  if (id >= ROWS*DI_) return;
  int d = id % DI_;
  int m = id / DI_;
  float z = xz[(size_t)m*768 + 384 + d];
  float y = (ysc[id] + xc[id]*Dp[d]) * siluf(z);
  Abf[id] = (__bf16)y;
}

__global__ void addto_kernel(float* __restrict__ dst, const float* __restrict__ src, int n)
{
  int id = blockIdx.x * blockDim.x + threadIdx.x;
  if (id < n) dst[id] += src[id];
}

__global__ void add3_kernel(float* __restrict__ hx, const float* __restrict__ y3,
                            const float* __restrict__ mlp, int n)
{
  int id = blockIdx.x * blockDim.x + threadIdx.x;
  if (id < n) hx[id] = hx[id] + y3[id] + mlp[id];
}

// window partition (cyclic roll by -shift), f32 -> bf16
__global__ void window_part_kernel(const float* __restrict__ src, __bf16* __restrict__ dst, int shift)
{
  int id = blockIdx.x * blockDim.x + threadIdx.x;
  if (id >= ROWS*CH) return;
  int c = id % CH, m = id / CH;
  int l = m & 63, wid = m >> 6;
  int b = wid >> 6, wloc = wid & 63, wr = wloc >> 3, wc = wloc & 7;
  int ly = l >> 3, lx = l & 7;
  int gy = (wr*8 + ly + shift) & 63;
  int gx = (wc*8 + lx + shift) & 63;
  dst[id] = (__bf16)src[((size_t)b*NTOK + gy*64 + gx)*CH + c];
}

// un-window + roll back + residual: y3 = sc + o
__global__ void window_merge_kernel(const float* __restrict__ o, const float* __restrict__ sc,
                                    float* __restrict__ y3, int shift)
{
  int id = blockIdx.x * blockDim.x + threadIdx.x;
  if (id >= ROWS*CH) return;
  int c = id % CH, m = id / CH;
  int b = m >> 12, pos = m & 4095, gy = pos >> 6, gx = pos & 63;
  int ry = (gy - shift) & 63, rx = (gx - shift) & 63;
  int wr = ry >> 3, wc = rx >> 3, ly = ry & 7, lx = rx & 7;
  int wid = b*64 + wr*8 + wc, l = ly*8 + lx;
  y3[id] = sc[id] + o[((size_t)wid*64 + l)*CH + c];
}

__device__ __forceinline__ int regionOf(int r){ return r < 56 ? 0 : (r < 60 ? 1 : 2); }

// =====================================================================
// Window attention: one block per (window, head); 512 threads = 16 waves.
// Stage0: V^T staged to LDS as bf16 (contiguous K for P@V fragments)
// Phase1: 16 waves, one 16x16 score tile each (WMMA, K=32=HD)
// Phase2: softmax rows in LDS (rel-pos bias + shift mask applied in P1)
// Phase3: 8 waves, P@V tiles (WMMA, K=64), both fragments from LDS (ds ops)
// =====================================================================
__global__ void swin_attn_kernel(const float* __restrict__ qkv, const float* __restrict__ rpb,
                                 __bf16* __restrict__ obf, int shift)
{
  __shared__ float  sS[64*64];
  __shared__ __bf16 sA[64*64];
  __shared__ __bf16 sV[32*64];   // [n][k] = V^T
  int blk = blockIdx.x;
  int win = blk / NHEADS, head = blk % NHEADS;
  int tid = (int)threadIdx.x;
  int w = tid >> 5, lane = tid & 31, half = lane >> 4, r = lane & 15;
  const float* base = qkv + (size_t)win * 64 * 576;
  int wloc = win & 63, wr = wloc >> 3, wc = wloc & 7;

  // ---- stage V^T ----
  for (int idx = tid; idx < 32*64; idx += 512) {
    int n = idx >> 6, k = idx & 63;
    sV[idx] = (__bf16)base[(size_t)k*576 + 384 + head*HDIM + n];
  }

  // ---- phase 1: S = Q K^T * scale + bias (+mask) ----
  {
    int tm = w >> 2, tn = w & 3;
    const float* qrow = base + (size_t)(tm*16 + r)*576 +       head*HDIM + half*8;
    const float* krow = base + (size_t)(tn*16 + r)*576 + 192 + head*HDIM + half*8;
    union { v16bf v; __bf16 e[16]; } fa, fb;
#pragma unroll
    for (int e = 0; e < 8; e++) {
      fa.e[e] = (__bf16)qrow[e];  fa.e[e+8] = (__bf16)qrow[e+16];
      fb.e[e] = (__bf16)krow[e];  fb.e[e+8] = (__bf16)krow[e+16];
    }
    v8f acc = {};
    acc = __builtin_amdgcn_wmma_f32_16x16x32_bf16(false, fa.v, false, fb.v,
                                                  (short)0, acc, false, false);
    int m = tn*16 + r;
    int my = m >> 3, mx = m & 7;
    int regM = regionOf(wr*8 + my)*3 + regionOf(wc*8 + mx);
#pragma unroll
    for (int i = 0; i < 8; i++) {
      int l = tm*16 + i + half*8;
      int ly = l >> 3, lx = l & 7;
      int idx = (ly - my + 7)*15 + (lx - mx + 7);
      float s = acc[i] * 0.17677669529f + rpb[idx*NHEADS + head];
      if (shift > 0) {
        int regL = regionOf(wr*8 + ly)*3 + regionOf(wc*8 + lx);
        if (regL != regM) s += -100.0f;
      }
      sS[l*64 + m] = s;
    }
  }
  __syncthreads();

  // ---- phase 2: softmax per row ----
  if (tid < 64) {
    float* rowp = sS + tid*64;
    float mx = -1e30f;
    for (int j = 0; j < 64; j++) mx = fmaxf(mx, rowp[j]);
    float sum = 0.f;
    for (int j = 0; j < 64; j++) { float p = expf(rowp[j]-mx); rowp[j] = p; sum += p; }
    float inv = 1.f / sum;
    for (int j = 0; j < 64; j++) sA[tid*64 + j] = (__bf16)(rowp[j]*inv);
  }
  __syncthreads();

  // ---- phase 3: O = P V  (all fragments from LDS) ----
  if (w < 8) {
    int tm = w >> 1, tn = w & 1;
    int n = tn*16 + r;
    v8f acc = {};
    for (int k0 = 0; k0 < 64; k0 += 32) {
      union { v16bf v; uint4 u[2]; } fa, fb;
      const __bf16* ar = sA + (tm*16 + r)*64 + k0 + half*8;
      const __bf16* br = sV + n*64 + k0 + half*8;
      fa.u[0] = *(const uint4*)(ar);
      fa.u[1] = *(const uint4*)(ar + 16);
      fb.u[0] = *(const uint4*)(br);
      fb.u[1] = *(const uint4*)(br + 16);
      acc = __builtin_amdgcn_wmma_f32_16x16x32_bf16(false, fa.v, false, fb.v,
                                                    (short)0, acc, false, false);
    }
#pragma unroll
    for (int i = 0; i < 8; i++) {
      int gl = win*64 + tm*16 + i + half*8;
      int col = head*HDIM + tn*16 + r;
      obf[(size_t)gl*CH + col] = (__bf16)acc[i];
    }
  }
}

// fused = w*lt + (1-w)*ht
__global__ void fuse_kernel(const float* __restrict__ lt, const float* __restrict__ hx,
                            const float* __restrict__ fw, float* __restrict__ out)
{
  int id = blockIdx.x * blockDim.x + threadIdx.x;
  if (id >= ROWS*CH) return;
  float w = 1.f / (1.f + expf(-fw[0]));
  out[id] = w*lt[id] + (1.f - w)*hx[id];
}

// mean over tokens per (b,c)
__global__ void colmean_kernel(const float* __restrict__ fused, float* __restrict__ mean)
{
  int b = blockIdx.x / CH, c = blockIdx.x % CH;
  float s = 0.f;
  for (int t = threadIdx.x; t < NTOK; t += 256)
    s += fused[((size_t)b*NTOK + t)*CH + c];
  __shared__ float sm[256];
  sm[threadIdx.x] = s; __syncthreads();
  for (int st = 128; st; st >>= 1) {
    if ((int)threadIdx.x < st) sm[threadIdx.x] += sm[threadIdx.x + st];
    __syncthreads();
  }
  if (threadIdx.x == 0) mean[blockIdx.x] = sm[0] / (float)NTOK;
}

__global__ void aux_head_kernel(const float* __restrict__ mean,
                                const float* __restrict__ w1, const float* __restrict__ b1,
                                const float* __restrict__ w2, const float* __restrict__ b2,
                                float* __restrict__ aux2)
{
  __shared__ float a1[2*96];
  int tid = (int)threadIdx.x;
  if (tid < 192) {
    int b = tid / 96, j = tid % 96;
    float s = b1[j];
    for (int c = 0; c < CH; c++) s += mean[b*CH + c] * w1[j*CH + c];
    a1[tid] = geluf(s);
  }
  __syncthreads();
  if (tid < 6) {
    int b = tid / 3, k = tid % 3;
    float s = b2[k];
    for (int j = 0; j < 96; j++) s += a1[b*96 + j] * w2[k*96 + j];
    aux2[tid] = s;
  }
}

__global__ void aux_bcast_kernel(const float* __restrict__ aux2, float* __restrict__ out)
{
  int id = blockIdx.x * blockDim.x + threadIdx.x;
  if (id >= BATCH*3*IMGSZ*IMGSZ) return;
  out[id] = aux2[id >> 16];
}

// =====================================================================
// host orchestration
// =====================================================================
extern "C" void kernel_launch(void* const* d_in, const int* in_sizes, int n_in,
                              void* d_out, int out_size, void* d_ws, size_t ws_size,
                              hipStream_t stream)
{
  (void)in_sizes; (void)n_in; (void)out_size; (void)ws_size;
  auto fin = [&](int i) -> const float* { return (const float*)d_in[i]; };

  const float* x      = fin(0);
  const float* blur_w = fin(1);
  const float* fusion_w = fin(62);
  const float* aux_w1 = fin(63); const float* aux_b1 = fin(64);
  const float* aux_w2 = fin(65); const float* aux_b2 = fin(66);

  // ---- workspace carve ----
  float* F = (float*)d_ws;
  size_t o = 0;
  auto alloc = [&](size_t n) { float* p = F + o; o += n; return p; };
  float* low  = alloc((size_t)BATCH*3*IMGSZ*IMGSZ);
  float* high = alloc((size_t)BATCH*3*IMGSZ*IMGSZ);
  float* lt   = alloc((size_t)ROWS*CH);
  float* hx   = alloc((size_t)ROWS*CH);
  float* t0   = alloc((size_t)ROWS*768);
  float* t1   = alloc((size_t)ROWS*768);
  float* xc   = alloc((size_t)ROWS*DI_);
  float* ysc  = alloc((size_t)ROWS*DI_);
  float* dbl  = alloc((size_t)ROWS*48);
  float* ln1  = alloc((size_t)ROWS*CH);
  float* lnb  = alloc((size_t)ROWS*CH);
  float* y3   = alloc((size_t)ROWS*CH);
  float* mlp  = alloc((size_t)ROWS*CH);
  float* meanb= alloc(512);
  float* aux2 = alloc(16);
  __bf16* Abf = (__bf16*)(F + o); o += (size_t)ROWS*768/2;
  __bf16* Wbf = (__bf16*)(F + o); o += (size_t)768*768/2;
  __bf16* obf = (__bf16*)(F + o); o += (size_t)ROWS*CH/2;

  auto elw = [&](int n) { return dim3((n + 255) / 256); };
  auto gemm = [&](const __bf16* A, const __bf16* W, const float* bias, float* D,
                  int M, int N, int K, int act) {
    int tilesN = N / 16;
    int nStrips = (tilesN + 3) / 4;
    int blocks = (M / 16) * nStrips;
    gemm_bf16_wmma_kernel<<<blocks, 128, 0, stream>>>(A, W, bias, D, M, N, K, act, nStrips);
  };

  // 1. frequency split
  blur_split_kernel<<<elw(BATCH*3*IMGSZ*IMGSZ), 256, 0, stream>>>(x, blur_w, low, high);

  // 2. patch embed (low -> lt, high -> hx)
  for (int e = 0; e < 2; e++) {
    const float* img = e ? high : low;
    int p = e ? 6 : 2;
    float* dst = e ? hx : lt;
    im2col_embed_kernel<<<elw(ROWS*64), 256, 0, stream>>>(img, Abf);
    convert_pad_kernel<<<elw(CH*64), 256, 0, stream>>>(fin(p+0), 48, CH, 48, Wbf, CH, 64);
    gemm(Abf, Wbf, fin(p+1), t0, ROWS, CH, 64, 0);
    ln_rows_kernel<<<elw(ROWS*32), 256, 0, stream>>>(t0, fin(p+2), fin(p+3), dst, ROWS, CH);
  }

  // 3. mamba blocks on lt
  for (int blk = 0; blk < 2; blk++) {
    int p = 10 + blk*11;
    const float *mg = fin(p+0), *mb = fin(p+1), *in_w = fin(p+2), *conv_w = fin(p+3),
                *conv_b = fin(p+4), *xp_w = fin(p+5), *dt_w = fin(p+6), *dt_b = fin(p+7),
                *A_log = fin(p+8), *Dp = fin(p+9), *out_w = fin(p+10);
    ln_rows_kernel<<<elw(ROWS*32), 256, 0, stream>>>(lt, mg, mb, lnb, ROWS, CH);
    convert_pad_kernel<<<elw(ROWS*CH), 256, 0, stream>>>(lnb, CH, ROWS, CH, Abf, ROWS, CH);
    convert_pad_kernel<<<elw(768*CH), 256, 0, stream>>>(in_w, CH, 768, CH, Wbf, 768, CH);
    gemm(Abf, Wbf, nullptr, t0, ROWS, 768, CH, 0);                       // xz
    conv1d_silu_kernel<<<elw(ROWS*DI_), 256, 0, stream>>>(t0, conv_w, conv_b, xc);
    convert_pad_kernel<<<elw(ROWS*DI_), 256, 0, stream>>>(xc, DI_, ROWS, DI_, Abf, ROWS, DI_);
    convert_pad_kernel<<<elw(48*DI_), 256, 0, stream>>>(xp_w, DI_, 44, DI_, Wbf, 48, DI_);
    gemm(Abf, Wbf, nullptr, dbl, ROWS, 48, DI_, 0);                      // dbl = [dt|B|C]
    convert_pad_kernel<<<elw(ROWS*32), 256, 0, stream>>>(dbl, 48, ROWS, DTRK, Abf, ROWS, 32);
    convert_pad_kernel<<<elw(DI_*32), 256, 0, stream>>>(dt_w, DTRK, DI_, DTRK, Wbf, DI_, 32);
    gemm(Abf, Wbf, dt_b, t1, ROWS, DI_, 32, 2);                          // delta (softplus)
    mamba_scan_kernel<<<48, 256, 0, stream>>>(t1, dbl, xc, A_log, ysc);
    mamba_gate_kernel<<<elw(ROWS*DI_), 256, 0, stream>>>(ysc, xc, t0, Dp, Abf);
    convert_pad_kernel<<<elw(CH*DI_), 256, 0, stream>>>(out_w, DI_, CH, DI_, Wbf, CH, DI_);
    gemm(Abf, Wbf, nullptr, t1, ROWS, CH, DI_, 0);
    addto_kernel<<<elw(ROWS*CH), 256, 0, stream>>>(lt, t1, ROWS*CH);
  }

  // 4. swin blocks on hx
  for (int blk = 0; blk < 2; blk++) {
    int p = 32 + blk*15;
    int shift = blk ? (WSZ/2) : 0;
    const float *pre_g = fin(p+0), *pre_b = fin(p+1), *n1_g = fin(p+2), *n1_b = fin(p+3),
                *qkv_w = fin(p+4), *qkv_b = fin(p+5), *proj_w = fin(p+6), *proj_b = fin(p+7),
                *rpb = fin(p+8), *n2_g = fin(p+9), *n2_b = fin(p+10),
                *fc1_w = fin(p+11), *fc1_b = fin(p+12), *fc2_w = fin(p+13), *fc2_b = fin(p+14);

    ln_rows_kernel<<<elw(ROWS*32), 256, 0, stream>>>(hx, pre_g, pre_b, ln1, ROWS, CH);  // y
    ln_rows_kernel<<<elw(ROWS*32), 256, 0, stream>>>(ln1, n1_g, n1_b, lnb, ROWS, CH);   // y2
    window_part_kernel<<<elw(ROWS*CH), 256, 0, stream>>>(lnb, Abf, shift);
    convert_pad_kernel<<<elw(576*CH), 256, 0, stream>>>(qkv_w, CH, 576, CH, Wbf, 576, CH);
    gemm(Abf, Wbf, qkv_b, t0, ROWS, 576, CH, 0);                         // qkv
    swin_attn_kernel<<<BATCH*64*NHEADS, 512, 0, stream>>>(t0, rpb, obf, shift);
    convert_pad_kernel<<<elw(CH*CH), 256, 0, stream>>>(proj_w, CH, CH, CH, Wbf, CH, CH);
    gemm(obf, Wbf, proj_b, t1, ROWS, CH, CH, 0);                         // proj (windowed)
    window_merge_kernel<<<elw(ROWS*CH), 256, 0, stream>>>(t1, ln1, y3, shift); // y3 = y + o
    ln_rows_kernel<<<elw(ROWS*32), 256, 0, stream>>>(y3, n2_g, n2_b, lnb, ROWS, CH);
    convert_pad_kernel<<<elw(ROWS*CH), 256, 0, stream>>>(lnb, CH, ROWS, CH, Abf, ROWS, CH);
    convert_pad_kernel<<<elw(MLPH_*CH), 256, 0, stream>>>(fc1_w, CH, MLPH_, CH, Wbf, MLPH_, CH);
    gemm(Abf, Wbf, fc1_b, t0, ROWS, MLPH_, CH, 1);                       // gelu
    convert_pad_kernel<<<elw(ROWS*MLPH_), 256, 0, stream>>>(t0, MLPH_, ROWS, MLPH_, Abf, ROWS, MLPH_);
    convert_pad_kernel<<<elw(CH*MLPH_), 256, 0, stream>>>(fc2_w, MLPH_, CH, MLPH_, Wbf, CH, MLPH_);
    gemm(Abf, Wbf, fc2_b, mlp, ROWS, CH, MLPH_, 0);
    add3_kernel<<<elw(ROWS*CH), 256, 0, stream>>>(hx, y3, mlp, ROWS*CH); // hx += y3 + mlp
  }

  // 5. fusion + aux head
  float* outF = (float*)d_out;
  fuse_kernel<<<elw(ROWS*CH), 256, 0, stream>>>(lt, hx, fusion_w, outF);
  colmean_kernel<<<BATCH*CH, 256, 0, stream>>>(outF, meanb);
  aux_head_kernel<<<1, 256, 0, stream>>>(meanb, aux_w1, aux_b1, aux_w2, aux_b2, aux2);
  aux_bcast_kernel<<<elw(BATCH*3*IMGSZ*IMGSZ), 256, 0, stream>>>(aux2, outF + (size_t)ROWS*CH);
}
